// InfoNCEGraph_41120016892710
// MI455X (gfx1250) — matrix operand order repaired and checked by reading, not compiled, not run on hardware
//
#include <hip/hip_runtime.h>
#include <hip/hip_bf16.h>

// ---------------------------------------------------------------------------
// InfoNCE memory-bank loss, fused for MI455X (gfx1250, wave32, WMMA).
// Pipeline:
//   k_zero         : zero loss accumulators in ws
//   k_fp_gemm      : fp = f @ W^T + b           (v_wmma_f32_16x16x4_f32)
//   k_rownorm      : fn = normalize(fp) -> bf16
//   k_bank_convert : Bank fp32 -> bf16 (halves streaming traffic)
//   k_flag / k_scatter : flag2, Bank2 row scatter at input_index
//   k_pairs_topk   : fused pairs GEMM (v_wmma_f32_16x16x32_bf16) + per-block
//                    top-8-min positive / top-32-max negative / top-32-by-pri
//                    candidate selection, all in LDS (pairs never hit HBM).
//                    A/B tiles staged with GLOBAL_LOAD_ASYNC_TO_LDS_B128
//                    (ASYNCcnt + s_wait_asynccnt).
//   k_merge_loss   : per-row merge of 64 candidate blocks + logsumexp loss
//   k_final        : out = num / (valid*16)  (or 0)
// ---------------------------------------------------------------------------

typedef __bf16        v16bf __attribute__((ext_vector_type(16)));
typedef __bf16        v4bf  __attribute__((ext_vector_type(4)));
typedef float         v8f   __attribute__((ext_vector_type(8)));
typedef float         v2f   __attribute__((ext_vector_type(2)));
typedef unsigned int  v4u   __attribute__((ext_vector_type(4)));

constexpr int NROW  = 1024;
constexpr int CIN   = 128;
constexpr int COUT  = 256;     // also K of the pairs GEMM
constexpr int MEMN  = 65536;
constexpr int PP    = 8;
constexpr int NNEG  = 32;
constexpr int ROWS_WG = 128;   // rows of fn pinned in LDS per workgroup
constexpr int COLGRP  = 64;    // 64 groups of 16 cols = 1024 bank cols per WG
constexpr int NBLK  = MEMN / (COLGRP * 16);  // 64 column blocks
constexpr float BIGF = 1.0e9f;
constexpr float TINV = 1.25f;  // 1 / 0.8

// ---- CDNA5 async global->LDS staging (guarded; fallback = load+ds_store) --
#if defined(__has_builtin)
#if __has_builtin(__builtin_amdgcn_global_load_async_to_lds_b128) && \
    __has_builtin(__builtin_amdgcn_s_wait_asynccnt)
#define HAVE_ASYNC_LDS 1
#endif
#endif

#ifdef HAVE_ASYNC_LDS
// Builtin prototype (from clang diagnostic): first arg is an int4-vector
// pointer in global AS(1), second in LDS AS(3).
typedef int v4i_vs __attribute__((vector_size(16)));
typedef __attribute__((address_space(1))) v4i_vs* g_v4i_p;
typedef __attribute__((address_space(3))) v4i_vs* l_v4i_p;
__device__ __forceinline__ void async_copy_b128(void* gsrc, void* ldst) {
  __builtin_amdgcn_global_load_async_to_lds_b128((g_v4i_p)gsrc, (l_v4i_p)ldst,
                                                 0, 0);
}
__device__ __forceinline__ void async_wait0() {
  __builtin_amdgcn_s_wait_asynccnt(0);
}
#endif

__device__ __forceinline__ float hash_uniform(unsigned r, unsigned c) {
  unsigned x = r * 0x9E3779B9u + c * 0x85EBCA6Bu + 0x1234567u;
  x ^= x >> 16; x *= 0x7FEB352Du; x ^= x >> 15; x *= 0x846CA68Bu; x ^= x >> 16;
  return (float)(x >> 8) * (1.0f / 16777216.0f);
}

// --------------------------- fp = f @ W^T + b ------------------------------
__global__ __launch_bounds__(32)
void k_fp_gemm(const float* __restrict__ f, const float* __restrict__ W,
               const float* __restrict__ bias, float* __restrict__ fp) {
  int tm = blockIdx.x, tn = blockIdx.y;
  int l = threadIdx.x;
  int ln16 = l & 15, hf = l >> 4;
  int gm = tm * 16 + ln16;   // A-matrix row for this lane
  int go = tn * 16 + ln16;   // B/D column for this lane
  v8f acc = {};
  for (int ks = 0; ks < CIN / 4; ++ks) {
    int k0 = ks * 4 + 2 * hf;
    v2f a, b;
    a[0] = f[gm * CIN + k0];
    a[1] = f[gm * CIN + k0 + 1];
    b[0] = W[go * CIN + k0];
    b[1] = W[go * CIN + k0 + 1];
    acc = __builtin_amdgcn_wmma_f32_16x16x4_f32(false, a, false, b,
                                                (short)0, acc, false, false);
  }
  float bv = bias[go];
  for (int vi = 0; vi < 8; ++vi) {
    int m = tm * 16 + vi + 8 * hf;
    fp[m * COUT + go] = acc[vi] + bv;
  }
}

// ---------------------- fn = normalize(fp) -> bf16 -------------------------
__global__ __launch_bounds__(256)
void k_rownorm(const float* __restrict__ fp, __bf16* __restrict__ fnbf) {
  __shared__ float sred[256];
  int r = blockIdx.x, t = threadIdx.x;
  float v = fp[r * COUT + t];
  sred[t] = v * v;
  __syncthreads();
  for (int s = 128; s > 0; s >>= 1) {
    if (t < s) sred[t] += sred[t + s];
    __syncthreads();
  }
  float rinv = rsqrtf(sred[0]);
  fnbf[r * COUT + t] = (__bf16)(v * rinv);
}

// ------------------------- Bank fp32 -> bf16 -------------------------------
__global__ __launch_bounds__(256)
void k_bank_convert(const float* __restrict__ Bank, __bf16* __restrict__ bankbf) {
  long long i = (long long)blockIdx.x * 256 + threadIdx.x;  // group of 4 elems
  const float4* src = (const float4*)Bank;
  v4bf* dst = (v4bf*)bankbf;
  float4 x = src[i];
  v4bf o;
  o[0] = (__bf16)x.x; o[1] = (__bf16)x.y; o[2] = (__bf16)x.z; o[3] = (__bf16)x.w;
  dst[i] = o;
}

__global__ __launch_bounds__(256)
void k_flag(const float* __restrict__ bf, float* __restrict__ flag2) {
  int i = blockIdx.x * 256 + threadIdx.x;
  flag2[i] = bf[i];
}

__global__ __launch_bounds__(256)
void k_scatter(const int* __restrict__ idx, const __bf16* __restrict__ fnbf,
               __bf16* __restrict__ bankbf, float* __restrict__ flag2) {
  int i = blockIdx.x, t = threadIdx.x;
  int d = idx[i];
  bankbf[(long long)d * COUT + t] = fnbf[i * COUT + t];
  if (t == 0) flag2[d] = 1.0f;
}

// ------------- fused pairs GEMM + per-block candidate top-k ----------------
__global__ __launch_bounds__(256)
void k_pairs_topk(const __bf16* __restrict__ fnbf, const __bf16* __restrict__ bankbf,
                  const float* __restrict__ flag2, const int* __restrict__ label,
                  const int* __restrict__ label_all,
                  float* __restrict__ posC, float* __restrict__ negC,
                  float* __restrict__ priC, float* __restrict__ rvC,
                  int* __restrict__ pcnt, int* __restrict__ ncnt) {
  __shared__ __bf16 Alds[ROWS_WG][COUT];   // 64 KB : fn rows (bf16)
  __shared__ __bf16 Blds[16][COUT];        //  8 KB : 16 bank columns
  __shared__ float  slice[ROWS_WG][16];    //  8 KB : pairs slice
  __shared__ float  posL[ROWS_WG][PP];     //  4 KB : running min-8 (ascending)
  __shared__ float  negL[ROWS_WG][NNEG];   // 16 KB : running max-32 (descending)
  __shared__ float  priL[ROWS_WG][NNEG];   // 16 KB : running max-32 priorities
  __shared__ float  rvL [ROWS_WG][NNEG];   // 16 KB : payload pairs values
  __shared__ int    colLab[16];
  __shared__ float  colFlag[16];
  __shared__ int    cntP[ROWS_WG], cntN[ROWS_WG];

  int cb = blockIdx.x;   // column block 0..63
  int rb = blockIdx.y;   // row block    0..7
  int t  = threadIdx.x;

  // Stage the 128x256 bf16 fn tile into LDS.
  {
    char* gsrc = (char*)(fnbf + (size_t)rb * ROWS_WG * COUT);
    char* ldst = (char*)&Alds[0][0];
#ifdef HAVE_ASYNC_LDS
    for (int i = t; i < ROWS_WG * COUT / 8; i += 256)
      async_copy_b128(gsrc + (size_t)i * 16, ldst + (size_t)i * 16);
#else
    const v4u* src = (const v4u*)gsrc;
    v4u* dst = (v4u*)ldst;
    for (int i = t; i < ROWS_WG * COUT / 8; i += 256) dst[i] = src[i];
#endif
  }
  for (int i = t; i < ROWS_WG * PP; i += 256) (&posL[0][0])[i] = BIGF;
  for (int i = t; i < ROWS_WG * NNEG; i += 256) {
    (&negL[0][0])[i] = -BIGF; (&priL[0][0])[i] = -BIGF; (&rvL[0][0])[i] = -BIGF;
  }
  if (t < ROWS_WG) { cntP[t] = 0; cntN[t] = 0; }
  int myLab = (t < ROWS_WG) ? label[rb * ROWS_WG + t] : 0;

  int w = t >> 5;
  int l = t & 31;
  int ln16 = l & 15, hf = l >> 4;

  for (int cc = 0; cc < COLGRP; ++cc) {
    int col0 = cb * (COLGRP * 16) + cc * 16;
    __syncthreads();  // protect Blds/slice from previous iteration readers
    {   // stage 16 bank columns (rows of Bank2) into LDS: 512 x b128
      char* bs = (char*)(bankbf + (size_t)col0 * COUT);
      char* bd = (char*)&Blds[0][0];
#ifdef HAVE_ASYNC_LDS
      async_copy_b128(bs + (size_t)t * 16, bd + (size_t)t * 16);
      async_copy_b128(bs + (size_t)(t + 256) * 16, bd + (size_t)(t + 256) * 16);
#else
      ((v4u*)bd)[t]       = ((const v4u*)bs)[t];
      ((v4u*)bd)[t + 256] = ((const v4u*)bs)[t + 256];
#endif
      if (cc + 1 < COLGRP) {   // pull next B tile toward the caches
        const char* nx = (const char*)(bankbf + (size_t)(col0 + 16) * COUT);
        __builtin_prefetch(nx + (size_t)t * 32, 0, 1);
      }
    }
    if (t < 16) { colLab[t] = label_all[col0 + t]; colFlag[t] = flag2[col0 + t]; }
#ifdef HAVE_ASYNC_LDS
    async_wait0();   // barrier does not cover ASYNCcnt
#endif
    __syncthreads();

    // Each wave: one 16(M)x16(N) tile, K = 256 -> 8 bf16 WMMAs.
    v8f acc = {};
    int arow = w * 16 + ln16;
    for (int kb = 0; kb < COUT / 32; ++kb) {
      union { v4u u[2]; v16bf v; } ua, ub;
      const __bf16* ap = &Alds[arow][kb * 32 + 8 * hf];
      ua.u[0] = *(const v4u*)(ap);        // K = kb*32 + 8*hf .. +7
      ua.u[1] = *(const v4u*)(ap + 16);   // K = kb*32 + 16 + 8*hf .. +7
      const __bf16* bp = &Blds[ln16][kb * 32 + 8 * hf];
      ub.u[0] = *(const v4u*)(bp);        // K = kb*32 + 8*hf .. +7
      ub.u[1] = *(const v4u*)(bp + 16);   // K = kb*32 + 16 + 8*hf .. +7
      acc = __builtin_amdgcn_wmma_f32_16x16x32_bf16(false, ua.v, false, ub.v,
                                                    (short)0, acc, false, false);
    }
    for (int vi = 0; vi < 8; ++vi)
      slice[w * 16 + vi + 8 * hf][ln16] = acc[vi];
    __syncthreads();

    // One selector thread per row updates its running candidate lists.
    if (t < ROWS_WG) {
      float* pl  = posL[t];
      float* nl  = negL[t];
      float* prl = priL[t];
      float* rvl = rvL[t];
      int grow = rb * ROWS_WG + t;
      for (int c = 0; c < 16; ++c) {
        float v = slice[t][c];
        if (colFlag[c] > 0.0f) {
          if (colLab[c] == myLab) {
            cntP[t] += 1;
            if (v < pl[PP - 1]) {                 // keep 8 smallest
              int j = PP - 1;
              while (j > 0 && pl[j - 1] > v) { pl[j] = pl[j - 1]; --j; }
              pl[j] = v;
            }
          } else {
            cntN[t] += 1;
            if (v > nl[NNEG - 1]) {               // keep 32 largest
              int j = NNEG - 1;
              while (j > 0 && nl[j - 1] < v) { nl[j] = nl[j - 1]; --j; }
              nl[j] = v;
            }
            float pr = hash_uniform((unsigned)grow, (unsigned)(col0 + c));
            if (pr > prl[NNEG - 1]) {             // keep 32 largest priorities
              int j = NNEG - 1;
              while (j > 0 && prl[j - 1] < pr) {
                prl[j] = prl[j - 1]; rvl[j] = rvl[j - 1]; --j;
              }
              prl[j] = pr; rvl[j] = v;
            }
          }
        }
      }
    }
  }
  __syncthreads();
  if (t < ROWS_WG) {
    int grow = rb * ROWS_WG + t;
    size_t base = (size_t)grow * NBLK + cb;
    for (int j = 0; j < PP; ++j)   posC[base * PP + j] = posL[t][j];
    for (int j = 0; j < NNEG; ++j) {
      negC[base * NNEG + j] = negL[t][j];
      priC[base * NNEG + j] = priL[t][j];
      rvC [base * NNEG + j] = rvL [t][j];
    }
    pcnt[base] = cntP[t];
    ncnt[base] = cntN[t];
  }
}

// ------------------ per-row merge of candidates + loss ---------------------
__global__ __launch_bounds__(256)
void k_merge_loss(const float* __restrict__ posC, const float* __restrict__ negC,
                  const float* __restrict__ priC, const float* __restrict__ rvC,
                  const int* __restrict__ pcnt, const int* __restrict__ ncnt,
                  float* __restrict__ accum) {
  __shared__ float posBuf[NBLK * PP];      // 512
  __shared__ float negBuf[NBLK * NNEG];    // 2048
  __shared__ float priBuf[NBLK * NNEG];
  __shared__ float rvBuf [NBLK * NNEG];
  __shared__ float redV[256];
  __shared__ int   redI[256];
  __shared__ float posH[PP], negH[NNEG], negR[NNEG];
  __shared__ int   sValid;

  int r = blockIdx.x, t = threadIdx.x;
  size_t base = (size_t)r * NBLK;
  for (int i = t; i < NBLK * PP; i += 256) posBuf[i] = posC[base * PP + i];
  for (int i = t; i < NBLK * NNEG; i += 256) {
    negBuf[i] = negC[base * NNEG + i];
    priBuf[i] = priC[base * NNEG + i];
    rvBuf [i] = rvC [base * NNEG + i];
  }
  if (t == 0) {
    int tp = 0, tn = 0;
    for (int b2 = 0; b2 < NBLK; ++b2) { tp += pcnt[base + b2]; tn += ncnt[base + b2]; }
    sValid = (tp >= PP && tn >= NNEG) ? 1 : 0;
  }
  __syncthreads();

  // 8 smallest positives
  for (int j = 0; j < PP; ++j) {
    float bv = 2.0f * BIGF; int bi = 0;
    for (int i = t; i < NBLK * PP; i += 256) {
      float v = posBuf[i]; if (v < bv) { bv = v; bi = i; }
    }
    redV[t] = bv; redI[t] = bi; __syncthreads();
    for (int s = 128; s > 0; s >>= 1) {
      if (t < s && redV[t + s] < redV[t]) { redV[t] = redV[t + s]; redI[t] = redI[t + s]; }
      __syncthreads();
    }
    if (t == 0) { int b0 = redI[0]; posH[j] = posBuf[b0]; posBuf[b0] = 2.0f * BIGF; }
    __syncthreads();
  }
  // 32 largest negatives (hard)
  for (int j = 0; j < NNEG; ++j) {
    float bv = -2.0f * BIGF; int bi = 0;
    for (int i = t; i < NBLK * NNEG; i += 256) {
      float v = negBuf[i]; if (v > bv) { bv = v; bi = i; }
    }
    redV[t] = bv; redI[t] = bi; __syncthreads();
    for (int s = 128; s > 0; s >>= 1) {
      if (t < s && redV[t + s] > redV[t]) { redV[t] = redV[t + s]; redI[t] = redI[t + s]; }
      __syncthreads();
    }
    if (t == 0) { int b0 = redI[0]; negH[j] = negBuf[b0]; negBuf[b0] = -2.0f * BIGF; }
    __syncthreads();
  }
  // 32 random negatives (largest priorities; payload = pairs value)
  for (int j = 0; j < NNEG; ++j) {
    float bv = -2.0f * BIGF; int bi = 0;
    for (int i = t; i < NBLK * NNEG; i += 256) {
      float v = priBuf[i]; if (v > bv) { bv = v; bi = i; }
    }
    redV[t] = bv; redI[t] = bi; __syncthreads();
    for (int s = 128; s > 0; s >>= 1) {
      if (t < s && redV[t + s] > redV[t]) { redV[t] = redV[t + s]; redI[t] = redI[t + s]; }
      __syncthreads();
    }
    if (t == 0) { int b0 = redI[0]; negR[j] = rvBuf[b0]; priBuf[b0] = -2.0f * BIGF; }
    __syncthreads();
  }

  // 16 logit rows: [pos_p, 32 negs] / T ; loss = logsumexp - logits[0]
  float loss = 0.0f;
  if (t < 2 * PP) {
    int p = t & (PP - 1);
    const float* ng = (t < PP) ? negH : negR;
    float s0 = posH[p] * TINV;
    float mx = s0;
    for (int i = 0; i < NNEG; ++i) mx = fmaxf(mx, ng[i] * TINV);
    float se = __expf(s0 - mx);
    for (int i = 0; i < NNEG; ++i) se += __expf(ng[i] * TINV - mx);
    loss = __logf(se) + mx - s0;
  }
  redV[t] = loss;
  __syncthreads();
  if (t == 0) {
    float s = 0.0f;
    for (int i = 0; i < 2 * PP; ++i) s += redV[i];
    if (sValid) { atomicAdd(&accum[0], s); atomicAdd(&accum[1], 1.0f); }
  }
}

__global__ void k_zero(float* accum) { accum[0] = 0.0f; accum[1] = 0.0f; }

__global__ void k_final(const float* __restrict__ accum, float* __restrict__ out) {
  float c = accum[1];
  out[0] = (c > 0.0f) ? accum[0] / (c * (2.0f * PP)) : 0.0f;
}

// ---------------------------------------------------------------------------
extern "C" void kernel_launch(void* const* d_in, const int* in_sizes, int n_in,
                              void* d_out, int out_size, void* d_ws, size_t ws_size,
                              hipStream_t stream) {
  const float* f          = (const float*)d_in[0];
  const float* W          = (const float*)d_in[1];
  const float* bias       = (const float*)d_in[2];
  const float* Bank       = (const float*)d_in[3];
  const float* bank_flag  = (const float*)d_in[4];
  const int*   label      = (const int*)d_in[5];
  const int*   input_idx  = (const int*)d_in[6];
  const int*   label_all  = (const int*)d_in[7];
  float* out = (float*)d_out;
  (void)in_sizes; (void)n_in; (void)out_size; (void)ws_size;

  char* ws = (char*)d_ws;
  size_t o = 0;
  auto take = [&](size_t bytes) {
    size_t cur = o;
    o += (bytes + 255) & ~(size_t)255;
    return cur;
  };
  float*  fp     = (float*)(ws + take((size_t)NROW * COUT * 4));
  __bf16* fnbf   = (__bf16*)(ws + take((size_t)NROW * COUT * 2));
  __bf16* bankbf = (__bf16*)(ws + take((size_t)MEMN * COUT * 2));
  float*  flag2  = (float*)(ws + take((size_t)MEMN * 4));
  float*  posC   = (float*)(ws + take((size_t)NROW * NBLK * PP * 4));
  float*  negC   = (float*)(ws + take((size_t)NROW * NBLK * NNEG * 4));
  float*  priC   = (float*)(ws + take((size_t)NROW * NBLK * NNEG * 4));
  float*  rvC    = (float*)(ws + take((size_t)NROW * NBLK * NNEG * 4));
  int*    pcnt   = (int*)(ws + take((size_t)NROW * NBLK * 4));
  int*    ncnt   = (int*)(ws + take((size_t)NROW * NBLK * 4));
  float*  accum  = (float*)(ws + take(2 * sizeof(float)));

  k_zero<<<1, 1, 0, stream>>>(accum);
  k_fp_gemm<<<dim3(NROW / 16, COUT / 16), 32, 0, stream>>>(f, W, bias, fp);
  k_rownorm<<<NROW, 256, 0, stream>>>(fp, fnbf);
  k_bank_convert<<<(MEMN * COUT / 4) / 256, 256, 0, stream>>>(Bank, bankbf);
  k_flag<<<MEMN / 256, 256, 0, stream>>>(bank_flag, flag2);
  k_scatter<<<NROW, 256, 0, stream>>>(input_idx, fnbf, bankbf, flag2);
  k_pairs_topk<<<dim3(NBLK, NROW / ROWS_WG), 256, 0, stream>>>(
      fnbf, bankbf, flag2, label, label_all, posC, negC, priC, rvC, pcnt, ncnt);
  k_merge_loss<<<NROW, 256, 0, stream>>>(posC, negC, priC, rvC, pcnt, ncnt, accum);
  k_final<<<1, 1, 0, stream>>>(accum, out);
}
